// SAE_62070867361842
// MI455X (gfx1250) — matrix operand order repaired and checked by reading, not compiled, not run on hardware
//
#include <hip/hip_runtime.h>

#define B_TOK 8192
#define D_INN 2048
#define N_LAT 16384
#define TOPK  32
#define NT    4                     // n-tiles (of 16 latents) per wave iteration

typedef _Float16 v16h __attribute__((ext_vector_type(16)));
typedef _Float16 v8h  __attribute__((ext_vector_type(8)));
typedef _Float16 v4h  __attribute__((ext_vector_type(4)));
typedef float    v8f  __attribute__((ext_vector_type(8)));
typedef float    v4f  __attribute__((ext_vector_type(4)));

// ---------------- prep kernels: f32 -> f16 into workspace ----------------

__global__ void sae_prep_x(const float* __restrict__ x, const float* __restrict__ b_dec,
                           _Float16* __restrict__ xh) {
  size_t i4 = ((size_t)blockIdx.x * blockDim.x + threadIdx.x) * 4;
  v4f xv = *(const v4f*)(x + i4);
  v4f bv = *(const v4f*)(b_dec + (i4 & (size_t)(D_INN - 1)));
  v4h h;
#pragma unroll
  for (int j = 0; j < 4; ++j) h[j] = (_Float16)(xv[j] - bv[j]);
  *(v4h*)(xh + i4) = h;
}

__global__ void sae_prep_w(const float* __restrict__ w, _Float16* __restrict__ wh) {
  size_t i4 = ((size_t)blockIdx.x * blockDim.x + threadIdx.x) * 4;
  v4f wv = *(const v4f*)(w + i4);
  v4h h;
#pragma unroll
  for (int j = 0; j < 4; ++j) h[j] = (_Float16)wv[j];
  *(v4h*)(wh + i4) = h;
}

// ---------------- fused encoder GEMM + top-k + decode ----------------

#define XPITCH 2056                          // 2048 + 8 halves pad -> bank spread
#define SM_X_BYTES   (16 * XPITCH * 2)       // 65792
#define SM_SCR_OFF   SM_X_BYTES
#define SM_SCR_BYTES (8 * 256 * 4)           // 8 waves * one 16x16 f32 tile = 8192
#define SM_LV_OFF    (SM_SCR_OFF + SM_SCR_BYTES)
#define SM_LV_BYTES  (8 * 16 * TOPK * 4)
#define SM_LI_OFF    (SM_LV_OFF + SM_LV_BYTES)
#define SM_LI_BYTES  (8 * 16 * TOPK * 4)
#define SM_TOTAL     (SM_LI_OFF + SM_LI_BYTES)  // 106752 bytes -> 3 WGs / WGP

__launch_bounds__(256)
__global__ void sae_fused(const _Float16* __restrict__ xh, const _Float16* __restrict__ wh,
                          const float* __restrict__ b_enc, const float* __restrict__ W_dec,
                          const float* __restrict__ b_dec, float* __restrict__ out) {
  extern __shared__ char smem[];
  _Float16* lds_x   = (_Float16*)smem;
  float*    scr_all = (float*)(smem + SM_SCR_OFF);
  float*    lv_all  = (float*)(smem + SM_LV_OFF);
  unsigned* li_all  = (unsigned*)(smem + SM_LI_OFF);

  const int tid  = threadIdx.x;
  const int lane = tid & 31;
  const int wv   = tid >> 5;       // wave id 0..7
  const int half = lane >> 4;      // hi/lo half of wave
  const int mlo  = lane & 15;
  const int wg   = blockIdx.x;     // 16 token rows per workgroup

  // ---- stage x tile (16 rows x 2048 f16) into LDS with padded pitch ----
  {
    const v8h* src = (const v8h*)(xh + (size_t)wg * 16 * D_INN);
    for (int c = tid; c < 16 * (D_INN / 8); c += 256) {
      int row = c >> 8;            // 256 8-half chunks per row
      int col8 = c & 255;
      v8h v = src[c];
      *(v8h*)(lds_x + row * XPITCH + col8 * 8) = v;
    }
  }
  // ---- init per-wave top-k lists ----
  for (int e = tid; e < 8 * 16 * TOPK; e += 256) {
    lv_all[e] = -__builtin_inff();
    li_all[e] = 0u;
  }
  __syncthreads();

  float*    scr = scr_all + wv * 256;              // one 16x16 f32 tile per wave
  float*    lv  = lv_all + wv * 16 * TOPK;
  unsigned* li  = li_all + wv * 16 * TOPK;

  float curmin = -__builtin_inff();   // running min of this lane's row list (lanes 0..15)
  int   minpos = 0;

  const _Float16* arow = lds_x + mlo * XPITCH;   // A: row m = lane%16

  // ---- encoder GEMM: 4 n-tiles (64 latents) per wave iteration ----
  for (int tb = wv; tb < N_LAT / (NT * 16); tb += 8) {
    const int n0 = tb * (NT * 16);
    // B fragment base: column n = lane%16, K = half*16 + j contiguous along k.
    // Sub-tiles u=0..3 sit at +u*16 rows of W => +u*65536 bytes (24-bit ioffset).
    const _Float16* wrow = wh + (size_t)(n0 + mlo) * D_INN + half * 16;
#pragma unroll
    for (int u = 0; u < NT; ++u)
      __builtin_prefetch((const void*)(wrow + ((size_t)8 * NT * 16 + u * 16) * D_INN), 0, 1);

    v8f acc[NT];
#pragma unroll
    for (int u = 0; u < NT; ++u) acc[u] = (v8f){};

#pragma unroll 2
    for (int k0 = 0; k0 < D_INN; k0 += 32) {
      // A layout (16-bit 16x32): per lane, K contiguous [half*16, +16) to match B
      v8h alo = *(const v8h*)(arow + k0 + half * 16);
      v8h ahi = *(const v8h*)(arow + k0 + half * 16 + 8);
      v16h a = __builtin_shufflevector(alo, ahi, 0, 1, 2, 3, 4, 5, 6, 7,
                                       8, 9, 10, 11, 12, 13, 14, 15);
#pragma unroll
      for (int u = 0; u < NT; ++u) {
        v16h b = *(const v16h*)(wrow + (size_t)u * 16 * D_INN + k0);
        acc[u] = __builtin_amdgcn_wmma_f32_16x16x32_f16(false, a, false, b,
                                                        (short)0, acc[u], false, false);
      }
    }

    // ---- per sub-tile: bias add, spill 16x16 to LDS, streaming top-32 update ----
#pragma unroll
    for (int u = 0; u < NT; ++u) {
      // D element (VGPR r, lane L): n = L%16, m = r + 8*(L/16)
      float be = b_enc[n0 + u * 16 + mlo];
#pragma unroll
      for (int r = 0; r < 8; ++r)
        scr[(r + half * 8) * 16 + mlo] = acc[u][r] + be;
      asm volatile("s_wait_dscnt 0" ::: "memory");   // wave-internal LDS visibility
      __builtin_amdgcn_wave_barrier();

      if (lane < 16) {                 // lane m owns row m
        float*    svals = lv + lane * TOPK;
        unsigned* sidx  = li + lane * TOPK;
        const float* cand = scr + lane * 16;
        for (int j = 0; j < 16; ++j) {
          float v = cand[j];
          if (v > curmin) {
            svals[minpos] = v;
            sidx[minpos]  = (unsigned)(n0 + u * 16 + j);
            float nm = svals[0]; int np = 0;
            for (int q = 1; q < TOPK; ++q) {
              float qv = svals[q];
              if (qv < nm) { nm = qv; np = q; }
            }
            curmin = nm; minpos = np;
          }
        }
      }
      __builtin_amdgcn_wave_barrier();
    }
  }

  __syncthreads();

  // ---- merge 8 partial lists into wave 0's list (final top-32 per row) ----
  if (wv == 0 && lane < 16) {
    float*    svals = lv_all + lane * TOPK;
    unsigned* sidx  = li_all + lane * TOPK;
    for (int w2 = 1; w2 < 8; ++w2) {
      const float*    ov = lv_all + (w2 * 16 + lane) * TOPK;
      const unsigned* oi = li_all + (w2 * 16 + lane) * TOPK;
      for (int q2 = 0; q2 < TOPK; ++q2) {
        float v = ov[q2];
        if (v > curmin) {
          svals[minpos] = v;
          sidx[minpos]  = oi[q2];
          float nm = svals[0]; int np = 0;
          for (int q = 1; q < TOPK; ++q) {
            float qv = svals[q];
            if (qv < nm) { nm = qv; np = q; }
          }
          curmin = nm; minpos = np;
        }
      }
    }
  }
  __syncthreads();

  // ---- decode: 2 rows per wave, lanes sweep 32 contiguous cols of W_dec ----
#pragma unroll
  for (int rr = 0; rr < 2; ++rr) {
    const int m = wv * 2 + rr;
    float av[TOPK]; unsigned ai[TOPK];
#pragma unroll
    for (int q = 0; q < TOPK; ++q) {
      av[q] = lv_all[m * TOPK + q];
      ai[q] = li_all[m * TOPK + q];
    }
    float* orow = out + ((size_t)wg * 16 + m) * D_INN;
    for (int c = lane; c < D_INN; c += 32) {
      float a2 = b_dec[c];
#pragma unroll
      for (int q = 0; q < TOPK; ++q)
        a2 += av[q] * W_dec[(size_t)ai[q] * D_INN + c];
      orow[c] = a2;
    }
  }
}

// ---------------- launch ----------------

extern "C" void kernel_launch(void* const* d_in, const int* in_sizes, int n_in,
                              void* d_out, int out_size, void* d_ws, size_t ws_size,
                              hipStream_t stream) {
  const float* x     = (const float*)d_in[0];
  const float* W_enc = (const float*)d_in[1];
  const float* b_enc = (const float*)d_in[2];
  const float* W_dec = (const float*)d_in[3];
  const float* b_dec = (const float*)d_in[4];
  float* out = (float*)d_out;

  _Float16* xh = (_Float16*)d_ws;                    // [B_TOK][D_INN] f16 (32 MB)
  _Float16* wh = xh + (size_t)B_TOK * D_INN;         // [N_LAT][D_INN] f16 (64 MB)

  sae_prep_x<<<(B_TOK * (D_INN / 4)) / 256, 256, 0, stream>>>(x, b_dec, xh);
  sae_prep_w<<<(N_LAT * (D_INN / 4)) / 256, 256, 0, stream>>>(W_enc, wh);
  sae_fused<<<B_TOK / 16, 256, SM_TOTAL, stream>>>(xh, wh, b_enc, W_dec, b_dec, out);
}
